// RelPositionMultiHeadAttention2_44822278701361
// MI455X (gfx1250) — matrix operand order repaired and checked by reading, not compiled, not run on hardware
//
#include <hip/hip_runtime.h>

// ---------------------------------------------------------------------------
// RelPositionMultiHeadAttention (Transformer-XL style) for gfx1250 / MI455X.
// bf16 WMMA everywhere (v_wmma_f32_16x16x32_bf16), f32 accumulation,
// flash-style online softmax, rel_shift folded into a bijective scatter.
// GEMM A-panel staged via the Tensor Data Mover (tensor_load_to_lds).
// ---------------------------------------------------------------------------

typedef __attribute__((ext_vector_type(16))) __bf16        v16bf;
typedef __attribute__((ext_vector_type(8)))  float         v8f;
typedef __attribute__((ext_vector_type(4)))  unsigned int  v4u;
typedef __attribute__((ext_vector_type(8)))  int           i32x8;
typedef __attribute__((ext_vector_type(4)))  int           i32x4;

union BFA { v4u q[2]; v16bf v; };   // 32B: one WMMA 16-bit operand per lane

constexpr int Bc = 4, Sc = 1024, Dc = 1024, Hc = 16, DHc = 64;
constexpr int BHc = Bc * Hc;

__device__ __forceinline__ unsigned short f2bf(float f) {
  union { float f; unsigned int u; } x; x.f = f;
  return (unsigned short)((x.u + 0x7FFFu + ((x.u >> 16) & 1u)) >> 16); // RNE
}
__device__ __forceinline__ float bf2f(unsigned short h) {
  union { unsigned int u; float f; } x; x.u = ((unsigned int)h) << 16;
  return x.f;
}

// ------------------------------------------------------------------- TDM
// 2-D tile DMA: global (row-major, stride0 elems between rows) -> LDS packed.
// D# layout per CDNA5 ISA. data_size = 2 bytes.
#define ATHENA_HAS_TDM __has_builtin(__builtin_amdgcn_tensor_load_to_lds)

#if ATHENA_HAS_TDM
__device__ __forceinline__ void tdm_load_tile_2d(
    unsigned lds_addr, const void* gaddr, int tile_d0 /*row elems*/,
    int tile_d1 /*rows*/, int tensor_d0, int tensor_d1,
    unsigned long long stride0 /*elems*/) {
  const unsigned long long ga = (unsigned long long)(size_t)gaddr;
  v4u g0;
  g0.x = 1u;                          // count=1, user descriptor, no gather
  g0.y = lds_addr;                    // LDS byte address (tile packed)
  g0.z = (unsigned)ga;                // global_addr[31:0]
  g0.w = (unsigned)((ga >> 32) & 0x01FFFFFFull) | (2u << 30); // type=2 image
  i32x8 g1 = (i32x8)(0);
  g1[0] = 1 << 16;                    // workgroup_mask=0, data_size=1 (2B)
  g1[1] = (int)(((unsigned)tensor_d0 & 0xFFFFu) << 16);        // dim0 lo16
  g1[2] = (int)(((unsigned)tensor_d0 >> 16) |
                (((unsigned)tensor_d1 & 0xFFFFu) << 16));      // dim0 hi|dim1 lo
  g1[3] = (int)(((unsigned)tensor_d1 >> 16) |
                ((unsigned)tile_d0 << 16));                    // dim1 hi|tile0
  g1[4] = tile_d1;                    // tile_dim1 (tile_dim2 = 0)
  g1[5] = (int)(unsigned)(stride0 & 0xFFFFFFFFull);            // stride0 lo32
  g1[6] = (int)(unsigned)((stride0 >> 32) & 0xFFFFull);        // stride0 hi16
  g1[7] = 0;
  const i32x4 z4 = (i32x4)(0);
#if defined(__clang_major__) && (__clang_major__ >= 23)
  const i32x8 z8 = (i32x8)(0);
  __builtin_amdgcn_tensor_load_to_lds(g0, g1, z4, z4, z8, 0);
#else
  __builtin_amdgcn_tensor_load_to_lds(g0, g1, z4, z4, 0);
#endif
}
__device__ __forceinline__ void tdm_wait0() {
#if __has_builtin(__builtin_amdgcn_s_wait_tensorcnt)
  __builtin_amdgcn_s_wait_tensorcnt(0);
#else
  asm volatile("s_wait_tensorcnt 0x0" ::: "memory");
#endif
}
#endif // ATHENA_HAS_TDM

// ---------------------------------------------------------------- converts
__global__ void k_f32_to_bf16(const float* __restrict__ in,
                              unsigned short* __restrict__ out, int n) {
  for (int i = blockIdx.x * blockDim.x + threadIdx.x; i < n;
       i += gridDim.x * blockDim.x)
    out[i] = f2bf(in[i]);
}

// ------------------------------------------------- tiled bf16 WMMA GEMM
// C(MxN) = A(MxK) @ B(KxN); row-major; 64x64 block per WG (4 waves),
// each wave owns a 16-row stripe, 4 N-tiles, K stepped by 32.
// A panel staged by TDM (wave 0 issues the descriptor); B panel staged +
// transposed by the whole workgroup.
template <typename OutT>
__global__ __launch_bounds__(128) void k_gemm_bf16(
    const unsigned short* __restrict__ A, const unsigned short* __restrict__ B,
    OutT* __restrict__ C, int M, int N, int K) {
  __shared__ unsigned short As[64 * 32];   // A block, row-major [m][k]
  __shared__ unsigned short Bt[64 * 32];   // B block transposed [n][k]
  const int tid  = threadIdx.x;
  const int wave = tid >> 5, lane = tid & 31;
  const int ln   = lane & 15, half = lane >> 4;
  const int n0   = blockIdx.x * 64;
  const int m0   = blockIdx.y * 64;

  v8f acc[4];
  #pragma unroll
  for (int t = 0; t < 4; ++t) acc[t] = (v8f)(0.0f);

  for (int kt = 0; kt < K; kt += 32) {
#if ATHENA_HAS_TDM
    if (wave == 0) {  // one DMA for the whole 64x32 A tile
      tdm_load_tile_2d((unsigned)(size_t)(void*)As,
                       A + (size_t)m0 * K + kt,
                       /*tile row elems*/ 32, /*rows*/ 64,
                       /*tensor_d0*/ K, /*tensor_d1*/ M,
                       /*stride0*/ (unsigned long long)K);
      tdm_wait0();
    }
#else
    { // stage A 64x32 (coalesced b128 loads -> b128 LDS stores)
      const int r = tid >> 1, cc = (tid & 1) << 4;
      const unsigned short* src = A + (size_t)(m0 + r) * K + kt + cc;
      v4u d0 = *(const v4u*)src;
      v4u d1 = *(const v4u*)(src + 8);
      *(v4u*)&As[r * 32 + cc]     = d0;
      *(v4u*)&As[r * 32 + cc + 8] = d1;
    }
#endif
    { // stage B 32x64, transposed into Bt[n][k]
      const int kk2 = tid >> 2, nc = (tid & 3) << 4;
      const unsigned short* src = B + (size_t)(kt + kk2) * N + n0 + nc;
      v4u d0 = *(const v4u*)src;
      v4u d1 = *(const v4u*)(src + 8);
      if (kt + 32 < K) __builtin_prefetch(src + 32 * (size_t)N, 0, 0);
      unsigned short tmp[16];
      *(v4u*)&tmp[0] = d0; *(v4u*)&tmp[8] = d1;
      #pragma unroll
      for (int e = 0; e < 16; ++e) Bt[(nc + e) * 32 + kk2] = tmp[e];
    }
    __syncthreads();
    // A operand: lane ln = row m; halves split K per ISA 16-bit A layout
    BFA a;
    {
      const unsigned short* ap = &As[(wave * 16 + ln) * 32 + (half << 3)];
      a.q[0] = *(const v4u*)ap;        // K = kb .. kb+7
      a.q[1] = *(const v4u*)(ap + 16); // K = kb+16 .. kb+23
    }
    #pragma unroll
    for (int nt = 0; nt < 4; ++nt) {
      BFA b;
      const unsigned short* bp = &Bt[(nt * 16 + ln) * 32 + (half << 3)];
      b.q[0] = *(const v4u*)bp;        // K = kb .. kb+7
      b.q[1] = *(const v4u*)(bp + 16); // K = kb+16 .. kb+23
      acc[nt] = __builtin_amdgcn_wmma_f32_16x16x32_bf16(
          false, a.v, false, b.v, (short)0, acc[nt], false, false);
    }
    __syncthreads();
  }
  #pragma unroll
  for (int nt = 0; nt < 4; ++nt)
    #pragma unroll
    for (int r = 0; r < 8; ++r) {
      const int row = m0 + wave * 16 + r + (half << 3);
      const int col = n0 + nt * 16 + ln;
      if constexpr (sizeof(OutT) == 2)
        C[(size_t)row * N + col] = (OutT)f2bf(acc[nt][r]);
      else
        C[(size_t)row * N + col] = (OutT)acc[nt][r];
    }
}

// -------------------------------------- split heads + bias add + V transpose
__global__ void k_split_heads(const unsigned short* __restrict__ wh, // (B*S,3D)
                              const float* __restrict__ rwb,
                              const float* __restrict__ rrb,
                              unsigned short* __restrict__ qwo,  // (BH,S,DH)
                              unsigned short* __restrict__ qro,  // (BH,S,DH)
                              unsigned short* __restrict__ ko,   // (BH,S,DH)
                              unsigned short* __restrict__ vto) { // (BH,DH,S)
  const int idx = blockIdx.x * blockDim.x + threadIdx.x;
  if (idx >= Bc * Sc * Dc) return;
  const int d = idx & 63;
  const int h = (idx >> 6) & 15;
  const int s = (idx >> 10) & 1023;
  const int b = idx >> 20;
  const size_t rb = ((size_t)(b * Sc + s)) * (3 * Dc) + h * DHc + d;
  const float qv = bf2f(wh[rb]);
  const float kv = bf2f(wh[rb + Dc]);
  const float vv = bf2f(wh[rb + 2 * Dc]);
  const size_t bh = (size_t)b * Hc + h;
  const size_t o1 = (bh * Sc + s) * DHc + d;
  qwo[o1] = f2bf(qv + rwb[h * DHc + d]);
  qro[o1] = f2bf(qv + rrb[h * DHc + d]);
  ko[o1]  = f2bf(kv);
  vto[(bh * DHc + d) * Sc + s] = f2bf(vv);
}

__global__ void k_split_rk(const unsigned short* __restrict__ rkf, // (S,D)
                           unsigned short* __restrict__ rkh) {     // (H,S,DH)
  const int idx = blockIdx.x * blockDim.x + threadIdx.x;
  if (idx >= Sc * Dc) return;
  const int d = idx & 63;
  const int h = (idx >> 6) & 15;
  const int r = idx >> 10;
  rkh[((size_t)h * Sc + r) * DHc + d] = rkf[(size_t)r * Dc + h * DHc + d];
}

// ------------------------------- BD = (q+r_r_bias) @ r_k^T, shifted on store
__global__ void k_bd_zero(unsigned short* __restrict__ bd) { // row S-1 is zero
  const int idx = blockIdx.x * blockDim.x + threadIdx.x;
  if (idx >= BHc * Sc) return;
  const int bh = idx >> 10, j = idx & 1023;
  bd[(((size_t)bh * Sc) + (Sc - 1)) * Sc + j] = 0;
}

__global__ __launch_bounds__(128) void k_bd_scatter(
    const unsigned short* __restrict__ qr,  // (BH,S,DH)
    const unsigned short* __restrict__ rkh, // (H,S,DH)
    unsigned short* __restrict__ bd) {      // (BH,S,S) shifted
  const int tid = threadIdx.x, wave = tid >> 5, lane = tid & 31;
  const int ln = lane & 15, half = lane >> 4;
  const int bh = blockIdx.y, h = bh & 15;
  const int t  = blockIdx.x * 4 + wave;        // 16x16 raw tile per wave
  const int i0 = (t >> 6) << 4;
  const int r0 = (t & 63) << 4;
  v8f c = (v8f)(0.0f);
  #pragma unroll
  for (int cc = 0; cc < 2; ++cc) {             // K = DH in two 32-chunks
    BFA a, b;
    const unsigned short* ap =
        qr + ((size_t)bh * Sc + i0 + ln) * DHc + cc * 32 + (half << 3);
    a.q[0] = *(const v4u*)ap;
    a.q[1] = *(const v4u*)(ap + 16);
    const unsigned short* bp =
        rkh + ((size_t)h * Sc + r0 + ln) * DHc + cc * 32 + (half << 3);
    b.q[0] = *(const v4u*)bp;
    b.q[1] = *(const v4u*)(bp + 16);
    c = __builtin_amdgcn_wmma_f32_16x16x32_bf16(
        false, a.v, false, b.v, (short)0, c, false, false);
  }
  // rel_shift as bijective scatter: raw (i,r) -> out[i, i+S-1-r]  (r>=i)
  //                                 raw (i,r) -> out[i-1, i-2-r]  (r<=i-2)
  #pragma unroll
  for (int r = 0; r < 8; ++r) {
    const int i  = i0 + r + (half << 3);
    const int rr = r0 + ln;
    const unsigned short v = f2bf(c[r]);
    if (rr >= i)
      bd[((size_t)bh * Sc + i) * Sc + (i + Sc - 1 - rr)] = v;
    else if (rr <= i - 2)
      bd[((size_t)bh * Sc + (i - 1)) * Sc + (i - 2 - rr)] = v;
  }
}

// ---------------------------------------------- flash attention per i-tile
__global__ __launch_bounds__(128) void k_attn(
    const unsigned short* __restrict__ qw,   // (BH,S,DH) q + r_w_bias
    const unsigned short* __restrict__ kmat, // (BH,S,DH)
    const unsigned short* __restrict__ vT,   // (BH,DH,S)
    const unsigned short* __restrict__ bd,   // (BH,S,S) shifted BD
    const unsigned char*  __restrict__ mask, // (B,S,S) bool
    unsigned short* __restrict__ attnv) {    // (B,S,D)
  __shared__ unsigned short Pst[4][16 * 32]; // per-wave P staging (C->A layout)
  const int tid = threadIdx.x, wave = tid >> 5, lane = tid & 31;
  const int ln = lane & 15, half = lane >> 4;
  const int bh = blockIdx.y;
  const int b = bh >> 4, h = bh & 15;
  const int i0 = (blockIdx.x * 4 + wave) << 4;
  const float MINV = -3.402823466e38f;

  BFA aq[2]; // q tile 16 x 64 as two K=32 A-operands
  {
    const unsigned short* base = qw + ((size_t)bh * Sc + i0 + ln) * DHc;
    const int kb = half << 3;
    #pragma unroll
    for (int c = 0; c < 2; ++c) {
      aq[c].q[0] = *(const v4u*)(base + c * 32 + kb);
      aq[c].q[1] = *(const v4u*)(base + c * 32 + kb + 16);
    }
  }

  float mrow[8], lrow[8];
  #pragma unroll
  for (int r = 0; r < 8; ++r) { mrow[r] = MINV; lrow[r] = 0.0f; }
  v8f o[4];
  #pragma unroll
  for (int t = 0; t < 4; ++t) o[t] = (v8f)(0.0f);

  unsigned short* P = &Pst[wave][0];

  for (int jc = 0; jc < Sc; jc += 32) {
    float sv[2][8];
    unsigned int keep[2] = {0u, 0u};
    float rmax[8];
    #pragma unroll
    for (int r = 0; r < 8; ++r) rmax[r] = MINV;

    #pragma unroll
    for (int sub = 0; sub < 2; ++sub) {
      const int j0 = jc + (sub << 4);
      v8f ac = (v8f)(0.0f);
      #pragma unroll
      for (int c = 0; c < 2; ++c) {            // AC over K = DH
        BFA bk;
        const unsigned short* kp =
            kmat + ((size_t)bh * Sc + j0 + ln) * DHc + c * 32 + (half << 3);
        bk.q[0] = *(const v4u*)kp;
        bk.q[1] = *(const v4u*)(kp + 16);
        ac = __builtin_amdgcn_wmma_f32_16x16x32_bf16(
            false, aq[c].v, false, bk.v, (short)0, ac, false, false);
      }
      #pragma unroll
      for (int r = 0; r < 8; ++r) {            // add BD, scale, mask
        const int i = i0 + r + (half << 3);
        const int j = j0 + ln;
        float s = (ac[r] + bf2f(bd[((size_t)bh * Sc + i) * Sc + j])) * 8.0f;
        const bool mk = mask[((size_t)b * Sc + i) * Sc + j] != 0;
        s = mk ? MINV : s;                     // branchless: EXEC stays full
        keep[sub] |= (mk ? 0u : 1u) << r;
        sv[sub][r] = s;
        rmax[r] = fmaxf(rmax[r], s);
      }
    }
    // per-row online softmax state (rows live in 16-lane halves)
    #pragma unroll
    for (int r = 0; r < 8; ++r) {
      float v = rmax[r];
      v = fmaxf(v, __shfl_xor(v, 1, 32));
      v = fmaxf(v, __shfl_xor(v, 2, 32));
      v = fmaxf(v, __shfl_xor(v, 4, 32));
      v = fmaxf(v, __shfl_xor(v, 8, 32));
      const float nm   = fmaxf(mrow[r], v);
      const float corr = __expf(mrow[r] - nm);
      mrow[r] = nm;
      lrow[r] *= corr;
      #pragma unroll
      for (int nt = 0; nt < 4; ++nt) o[nt][r] *= corr;
    }
    #pragma unroll
    for (int sub = 0; sub < 2; ++sub)
      #pragma unroll
      for (int r = 0; r < 8; ++r) {
        const float e = __expf(sv[sub][r] - mrow[r]);
        float t = e;                            // denom includes masked terms
        t += __shfl_xor(t, 1, 32);
        t += __shfl_xor(t, 2, 32);
        t += __shfl_xor(t, 4, 32);
        t += __shfl_xor(t, 8, 32);
        lrow[r] += t;
        const float pv = ((keep[sub] >> r) & 1u) ? e : 0.0f; // prob zeroed
        P[(r + (half << 3)) * 32 + (sub << 4) + ln] = f2bf(pv);
      }
    asm volatile("s_wait_dscnt 0x0" ::: "memory");
    BFA pa; // re-read P in A layout (16 x 32)
    {
      const int kb = half << 3;
      pa.q[0] = *(const v4u*)&P[ln * 32 + kb];
      pa.q[1] = *(const v4u*)&P[ln * 32 + kb + 16];
    }
    #pragma unroll
    for (int nt = 0; nt < 4; ++nt) {            // O += P @ V  (N = DH)
      BFA bv;
      const unsigned short* vp =
          vT + ((size_t)bh * DHc + nt * 16 + ln) * Sc + jc + (half << 3);
      bv.q[0] = *(const v4u*)vp;
      bv.q[1] = *(const v4u*)(vp + 16);
      o[nt] = __builtin_amdgcn_wmma_f32_16x16x32_bf16(
          false, pa.v, false, bv.v, (short)0, o[nt], false, false);
    }
    asm volatile("s_wait_dscnt 0x0" ::: "memory");
  }
  #pragma unroll
  for (int nt = 0; nt < 4; ++nt)
    #pragma unroll
    for (int r = 0; r < 8; ++r) {
      const int i = i0 + r + (half << 3);
      const float vv = o[nt][r] / lrow[r];
      attnv[((size_t)b * Sc + i) * Dc + h * DHc + nt * 16 + ln] = f2bf(vv);
    }
}

// ---------------------------------------------------------------- driver
extern "C" void kernel_launch(void* const* d_in, const int* in_sizes, int n_in,
                              void* d_out, int out_size, void* d_ws,
                              size_t ws_size, hipStream_t stream) {
  (void)in_sizes; (void)n_in; (void)out_size; (void)ws_size;
  const float* query         = (const float*)d_in[0];
  // d_in[1] (key) and d_in[2] (value) are unused by the reference.
  const unsigned char* mask  = (const unsigned char*)d_in[3]; // bool (B,S,S)
  const float* pos           = (const float*)d_in[4];
  const float* Wqkv          = (const float*)d_in[5];
  const float* Wr            = (const float*)d_in[6];
  const float* Wo            = (const float*)d_in[7];
  const float* rrb           = (const float*)d_in[8];
  const float* rwb           = (const float*)d_in[9];
  float* out                 = (float*)d_out;

  char* ws = (char*)d_ws;
  size_t off = 0;
  auto take = [&](size_t elems) -> unsigned short* {
    unsigned short* p = (unsigned short*)(ws + off);
    off += (elems * 2 + 255) & ~(size_t)255;
    return p;
  };
  unsigned short* qb     = take((size_t)Bc * Sc * Dc);
  unsigned short* wqkvb  = take((size_t)Dc * 3 * Dc);
  unsigned short* wrb    = take((size_t)Dc * Dc);
  unsigned short* wob    = take((size_t)Dc * Dc);
  unsigned short* posb   = take((size_t)Sc * Dc);
  unsigned short* wheads = take((size_t)Bc * Sc * 3 * Dc);
  unsigned short* rkf    = take((size_t)Sc * Dc);
  unsigned short* qw     = take((size_t)BHc * Sc * DHc);
  unsigned short* qr     = take((size_t)BHc * Sc * DHc);
  unsigned short* kk     = take((size_t)BHc * Sc * DHc);
  unsigned short* vT     = take((size_t)BHc * Sc * DHc);
  unsigned short* rkh    = take((size_t)Hc * Sc * DHc);
  unsigned short* bd     = take((size_t)BHc * Sc * Sc);
  unsigned short* attnv  = take((size_t)Bc * Sc * Dc);

  k_f32_to_bf16<<<2048, 256, 0, stream>>>(query, qb,   Bc * Sc * Dc);
  k_f32_to_bf16<<<2048, 256, 0, stream>>>(Wqkv,  wqkvb, Dc * 3 * Dc);
  k_f32_to_bf16<<<1024, 256, 0, stream>>>(Wr,    wrb,  Dc * Dc);
  k_f32_to_bf16<<<1024, 256, 0, stream>>>(Wo,    wob,  Dc * Dc);
  k_f32_to_bf16<<<1024, 256, 0, stream>>>(pos,   posb, Sc * Dc);

  k_gemm_bf16<unsigned short>
      <<<dim3(3 * Dc / 64, Bc * Sc / 64), 128, 0, stream>>>(
          qb, wqkvb, wheads, Bc * Sc, 3 * Dc, Dc);
  k_gemm_bf16<unsigned short>
      <<<dim3(Dc / 64, Sc / 64), 128, 0, stream>>>(
          posb, wrb, rkf, Sc, Dc, Dc);

  k_split_heads<<<(Bc * Sc * Dc + 255) / 256, 256, 0, stream>>>(
      wheads, rwb, rrb, qw, qr, kk, vT);
  k_split_rk<<<(Sc * Dc + 255) / 256, 256, 0, stream>>>(rkf, rkh);

  k_bd_zero<<<(BHc * Sc + 255) / 256, 256, 0, stream>>>(bd);
  k_bd_scatter<<<dim3((Sc / 16) * (Sc / 16) / 4, BHc), 128, 0, stream>>>(
      qr, rkh, bd);

  k_attn<<<dim3(Sc / 64, BHc), 128, 0, stream>>>(qw, kk, vT, bd, mask, attnv);

  k_gemm_bf16<float><<<dim3(Dc / 64, Bc * Sc / 64), 128, 0, stream>>>(
      attnv, wob, out, Bc * Sc, Dc, Dc);
}
